// WindowAttention_67662914781969
// MI455X (gfx1250) — matrix operand order repaired and checked by reading, not compiled, not run on hardware
//
#include <hip/hip_runtime.h>
#include <hip/hip_bf16.h>
#include <cstdint>

// ---------------- problem constants ----------------
#define NWIN   64
#define LTOK   343
#define LPAD   352            // 22 * 16 = 11 * 32
#define DIM    384
#define NHEAD  12
#define HEADD  32
#define NROW   (NWIN * LTOK)  // 21952 = 1372 * 16 = 686 * 32 exactly
#define MT_N   1372           // row tiles of 16
#define MT2_N  686            // row tiles of 32
#define CT_N   22             // col tiles of 16 over LPAD
#define SCALE  0.17677669529663689f   // 32^-0.5

typedef __bf16 bf16;
typedef __attribute__((ext_vector_type(16))) __bf16 v16bf;
typedef __attribute__((ext_vector_type(8)))  float  v8f;

union V16 { v16bf v; uint4 q[2]; unsigned short us[16]; };

static __device__ __forceinline__ bf16 f2bf(float f) {
    unsigned u = __builtin_bit_cast(unsigned, f);
    unsigned r = (u + 0x7FFFu + ((u >> 16) & 1u)) >> 16;   // round-to-nearest-even
    unsigned short h = (unsigned short)r;
    return __builtin_bit_cast(bf16, h);
}

static __device__ __forceinline__ v8f wmma_bf16(v16bf a, v16bf b, v8f c) {
    return __builtin_amdgcn_wmma_f32_16x16x32_bf16(false, a, false, b, (short)0, c,
                                                   false, false);
}

// ---------------- prep kernels ----------------
// convert activations to bf16 once (amortized over 18 N-tile passes of the QKV GEMM)
__global__ void k_prep_x(const float* __restrict__ x, bf16* __restrict__ xb, int n) {
    int i = blockIdx.x * 256 + threadIdx.x;
    if (i < n) xb[i] = f2bf(x[i]);
}

// transpose weights to bf16, column-major rows of length 384
__global__ void k_prep_w(const float* __restrict__ qkv_w, const float* __restrict__ proj_w,
                         bf16* __restrict__ wqkvT, bf16* __restrict__ wprojT) {
    int i = blockIdx.x * 256 + threadIdx.x;
    if (i < 1152 * 384) {
        int n = i / 384, k = i % 384;
        wqkvT[i] = f2bf(qkv_w[k * 1152 + n]);
    }
    if (i < 384 * 384) {
        int n = i / 384, k = i % 384;
        wprojT[i] = f2bf(proj_w[k * 384 + n]);
    }
}

// dense relative-position bias (nh, L, L)
__global__ void k_prep_bias(const float* __restrict__ table, const int* __restrict__ rel_idx,
                            float* __restrict__ bias) {
    int i = blockIdx.x * 256 + threadIdx.x;
    if (i < NHEAD * LTOK * LTOK) {
        int h = i / (LTOK * LTOK);
        int r = i % (LTOK * LTOK);
        bias[i] = table[rel_idx[r] * NHEAD + h];
    }
}

// zero the padded V^T buffer so pad K-rows contribute 0
__global__ void k_zero_vt(bf16* __restrict__ vT, int n) {
    int i = blockIdx.x * 256 + threadIdx.x;
    if (i < n) vT[i] = f2bf(0.0f);
}

// ---------------- QKV GEMM ----------------
// one wave computes a 32(M) x 64(N) tile of Xb(21952x384,bf16) @ Wqkv(384x1152):
// 8 WMMAs per 12 b128 loads per k-step (B operand shared by both M halves).
// writes scaled-Q / K as (win, h, l, 32) bf16 and V^T as (win, h, d, LPAD) bf16
__global__ __launch_bounds__(32)
void k_qkv_gemm(const bf16* __restrict__ xb, const bf16* __restrict__ wT,
                bf16* __restrict__ qb, bf16* __restrict__ kb, bf16* __restrict__ vT) {
    const int lane = threadIdx.x & 31;
    const int lr = lane & 15, hf = lane >> 4;
    const int m0 = blockIdx.x * 32, n0 = blockIdx.y * 64;
    const int row0 = m0 + lr;                      // A rows (always < NROW)
    const int row1 = m0 + 16 + lr;

    v8f acc0[4] = {}, acc1[4] = {};
    for (int k0 = 0; k0 < DIM; k0 += 32) {
        // A operands: bf16 rows, contiguous K stripe -> two b128 loads each
        const bf16* ar0 = xb + (size_t)row0 * DIM + k0;
        const bf16* ar1 = xb + (size_t)row1 * DIM + k0;
        V16 a0, a1;
        a0.q[0] = *(const uint4*)(ar0 + hf * 16);
        a0.q[1] = *(const uint4*)(ar0 + hf * 16 + 8);
        a1.q[0] = *(const uint4*)(ar1 + hf * 16);
        a1.q[1] = *(const uint4*)(ar1 + hf * 16 + 8);
        #pragma unroll
        for (int t = 0; t < 4; ++t) {
            const int col = n0 + t * 16 + lr;      // < 1152
            const bf16* wr = wT + (size_t)col * DIM + k0 + hf * 16;  // contiguous K stripe
            V16 b;
            b.q[0] = ((const uint4*)wr)[0];
            b.q[1] = ((const uint4*)wr)[1];
            acc0[t] = wmma_bf16(a0.v, b.v, acc0[t]);
            acc1[t] = wmma_bf16(a1.v, b.v, acc1[t]);
        }
    }
    // scatter C into attention-friendly layouts
    #pragma unroll
    for (int half_m = 0; half_m < 2; ++half_m) {
        #pragma unroll
        for (int t = 0; t < 4; ++t) {
            #pragma unroll
            for (int j = 0; j < 8; ++j) {
                const int gM = m0 + half_m * 16 + j + 8 * hf;
                const int gN = n0 + t * 16 + lr;
                const float val = half_m ? acc1[t][j] : acc0[t][j];
                const int which  = gN / DIM;           // 0=Q 1=K 2=V (block-uniform)
                const int within = gN % DIM;
                const int h = within >> 5, d = within & 31;
                const int win = gM / LTOK, tok = gM % LTOK;
                if (which == 2) {
                    vT[((size_t)((win * NHEAD + h) * HEADD + d)) * LPAD + tok] = f2bf(val);
                } else {
                    const size_t base = ((size_t)((win * NHEAD + h) * LTOK + tok)) * HEADD + d;
                    if (which == 0) qb[base] = f2bf(val * SCALE);
                    else            kb[base] = f2bf(val);
                }
            }
        }
    }
}

// ---------------- fused window attention ----------------
// one wave per (win, head, 16-row tile). S tile in LDS, two-pass softmax, PV from LDS.
__global__ __launch_bounds__(32)
void k_attn(const bf16* __restrict__ qb, const bf16* __restrict__ kb,
            const bf16* __restrict__ vT, const float* __restrict__ bias,
            const unsigned char* __restrict__ mask, bf16* __restrict__ opre) {
    __shared__ float S[16 * 353];      // odd stride -> conflict-free per-row scans
    __shared__ bf16  P[16 * LPAD];
    __shared__ float rowinv[16];

    const int lane = threadIdx.x & 31;
    const int lr = lane & 15, hf = lane >> 4;
    int bid = blockIdx.x;
    const int mt  = bid % CT_N;  bid /= CT_N;
    const int h   = bid % NHEAD; const int win = bid / NHEAD;
    const int m0  = mt * 16;
    const size_t wh = (size_t)(win * NHEAD + h);

    // Q operand (A, 16x32): lane lr -> row m0+lr, contiguous K stripe
    V16 aq;
    {
        const int qrow = m0 + lr;
        if (qrow < LTOK) {
            const bf16* qr = qb + (wh * LTOK + qrow) * HEADD;
            aq.q[0] = *(const uint4*)(qr + hf * 16);
            aq.q[1] = *(const uint4*)(qr + hf * 16 + 8);
        } else {
            aq.q[0] = make_uint4(0u, 0u, 0u, 0u);
            aq.q[1] = make_uint4(0u, 0u, 0u, 0u);
        }
    }

    // ---- scores: S = Q K^T + bias, masked ----
    for (int ct = 0; ct < CT_N; ++ct) {
        const int j0 = ct * 16;
        const int kcol = j0 + lr;                 // B column N=lr -> K row j0+lr
        V16 bk;
        if (kcol < LTOK) {
            const bf16* kr = kb + (wh * LTOK + kcol) * HEADD + hf * 16;
            bk.q[0] = ((const uint4*)kr)[0];
            bk.q[1] = ((const uint4*)kr)[1];
        } else {
            bk.q[0] = make_uint4(0u, 0u, 0u, 0u);
            bk.q[1] = make_uint4(0u, 0u, 0u, 0u);
        }
        v8f c = {};
        c = wmma_bf16(aq.v, bk.v, c);
        #pragma unroll
        for (int j = 0; j < 8; ++j) {
            const int rl = j + 8 * hf;
            const int grow = m0 + rl, gcol = j0 + lr;
            float s;
            if (grow < LTOK && gcol < LTOK) {
                s = c[j] + bias[((size_t)h * LTOK + grow) * LTOK + gcol];
                if (mask[((size_t)win * LTOK + grow) * LTOK + gcol]) s = -INFINITY;
            } else {
                s = -INFINITY;
            }
            S[rl * 353 + gcol] = s;
        }
    }
    __syncthreads();

    // ---- softmax: lane pair (lr, lr+16) shares row lr ----
    {
        const int srow = lr;
        const int cbeg = hf * 176, cend = cbeg + 176;   // 352 = 2*176
        float mx = -INFINITY;
        for (int cc = cbeg; cc < cend; ++cc) mx = fmaxf(mx, S[srow * 353 + cc]);
        mx = fmaxf(mx, __shfl_xor(mx, 16, 32));
        float sum = 0.0f;
        for (int cc = cbeg; cc < cend; ++cc) {
            const float p = __expf(S[srow * 353 + cc] - mx);
            P[srow * LPAD + cc] = f2bf(p);
            sum += p;
        }
        sum += __shfl_xor(sum, 16, 32);
        if (hf == 0) rowinv[srow] = 1.0f / sum;
    }
    __syncthreads();

    // ---- O = P V  (K over LPAD = 11 steps of 32) ----
    v8f o0 = {}, o1 = {};
    for (int kt = 0; kt < LPAD / 32; ++kt) {
        const int k0 = kt * 32;
        V16 ap;                                   // A from LDS P, contiguous K stripe
        const bf16* pr = &P[lr * LPAD + k0];
        ap.q[0] = *(const uint4*)(pr + hf * 16);
        ap.q[1] = *(const uint4*)(pr + hf * 16 + 8);
        const bf16* v0 = vT + (wh * HEADD + lr)      * LPAD + k0 + hf * 16;
        const bf16* v1 = vT + (wh * HEADD + 16 + lr) * LPAD + k0 + hf * 16;
        V16 b0, b1;
        b0.q[0] = ((const uint4*)v0)[0]; b0.q[1] = ((const uint4*)v0)[1];
        b1.q[0] = ((const uint4*)v1)[0]; b1.q[1] = ((const uint4*)v1)[1];
        o0 = wmma_bf16(ap.v, b0.v, o0);
        o1 = wmma_bf16(ap.v, b1.v, o1);
    }
    #pragma unroll
    for (int j = 0; j < 8; ++j) {
        const int rl = j + 8 * hf;
        const int gtok = m0 + rl;
        if (gtok < LTOK) {
            const float inv = rowinv[rl];
            const size_t ob = ((size_t)win * LTOK + gtok) * DIM + h * HEADD;
            opre[ob + lr]      = f2bf(o0[j] * inv);
            opre[ob + 16 + lr] = f2bf(o1[j] * inv);
        }
    }
}

// ---------------- output projection ----------------
// OUT(21952x384) = Opre(bf16) @ projW + proj_b, one wave = 32x64 tile
__global__ __launch_bounds__(32)
void k_proj_gemm(const bf16* __restrict__ opre, const bf16* __restrict__ wT,
                 const float* __restrict__ pb, float* __restrict__ out) {
    const int lane = threadIdx.x & 31;
    const int lr = lane & 15, hf = lane >> 4;
    const int m0 = blockIdx.x * 32, n0 = blockIdx.y * 64;
    const int row0 = m0 + lr;
    const int row1 = m0 + 16 + lr;

    v8f acc0[4] = {}, acc1[4] = {};
    for (int k0 = 0; k0 < DIM; k0 += 32) {
        const bf16* ar0 = opre + (size_t)row0 * DIM + k0;
        const bf16* ar1 = opre + (size_t)row1 * DIM + k0;
        V16 a0, a1;
        a0.q[0] = *(const uint4*)(ar0 + hf * 16);
        a0.q[1] = *(const uint4*)(ar0 + hf * 16 + 8);
        a1.q[0] = *(const uint4*)(ar1 + hf * 16);
        a1.q[1] = *(const uint4*)(ar1 + hf * 16 + 8);
        #pragma unroll
        for (int t = 0; t < 4; ++t) {
            const int col = n0 + t * 16 + lr;     // < 384
            const bf16* wr = wT + (size_t)col * DIM + k0 + hf * 16;
            V16 b;
            b.q[0] = ((const uint4*)wr)[0];
            b.q[1] = ((const uint4*)wr)[1];
            acc0[t] = wmma_bf16(a0.v, b.v, acc0[t]);
            acc1[t] = wmma_bf16(a1.v, b.v, acc1[t]);
        }
    }
    #pragma unroll
    for (int t = 0; t < 4; ++t) {
        const int gN = n0 + t * 16 + lr;
        const float bv = pb[gN];
        #pragma unroll
        for (int j = 0; j < 8; ++j) {
            const int gM0 = m0 + j + 8 * hf;
            out[(size_t)gM0 * DIM + gN]        = acc0[t][j] + bv;
            out[(size_t)(gM0 + 16) * DIM + gN] = acc1[t][j] + bv;
        }
    }
}

// ---------------- launch ----------------
extern "C" void kernel_launch(void* const* d_in, const int* in_sizes, int n_in,
                              void* d_out, int out_size, void* d_ws, size_t ws_size,
                              hipStream_t stream) {
    const float*         x      = (const float*)d_in[0];
    const float*         qkv_w  = (const float*)d_in[1];
    const float*         table  = (const float*)d_in[2];
    const float*         proj_w = (const float*)d_in[3];
    const float*         proj_b = (const float*)d_in[4];
    const unsigned char* mask   = (const unsigned char*)d_in[5];
    const int*           relidx = (const int*)d_in[6];
    float* out = (float*)d_out;

    // workspace carving (256-B aligned)
    char* ws = (char*)d_ws;
    size_t off = 0;
    auto carve = [&](size_t bytes) { char* p = ws + off; off = (off + bytes + 255) & ~(size_t)255; return p; };
    bf16*  xb     = (bf16*) carve((size_t)NROW * DIM * 2);
    bf16*  wqkvT  = (bf16*) carve((size_t)1152 * 384 * 2);
    bf16*  wprojT = (bf16*) carve((size_t)384 * 384 * 2);
    float* bias   = (float*)carve((size_t)NHEAD * LTOK * LTOK * 4);
    bf16*  qb     = (bf16*) carve((size_t)NWIN * NHEAD * LTOK * HEADD * 2);
    bf16*  kb     = (bf16*) carve((size_t)NWIN * NHEAD * LTOK * HEADD * 2);
    bf16*  vT     = (bf16*) carve((size_t)NWIN * NHEAD * HEADD * LPAD * 2);
    bf16*  opre   = (bf16*) carve((size_t)NROW * DIM * 2);

    // prep
    {
        const int nx = NROW * DIM;
        k_prep_x<<<(nx + 255) / 256, 256, 0, stream>>>(x, xb, nx);
    }
    k_prep_w   <<<(1152 * 384 + 255) / 256, 256, 0, stream>>>(qkv_w, proj_w, wqkvT, wprojT);
    k_prep_bias<<<(NHEAD * LTOK * LTOK + 255) / 256, 256, 0, stream>>>(table, relidx, bias);
    {
        const int nvt = NWIN * NHEAD * HEADD * LPAD;
        k_zero_vt<<<(nvt + 255) / 256, 256, 0, stream>>>(vT, nvt);
    }
    // QKV projection
    k_qkv_gemm<<<dim3(MT2_N, 1152 / 64), 32, 0, stream>>>(xb, wqkvT, qb, kb, vT);
    // fused attention
    k_attn<<<NWIN * NHEAD * CT_N, 32, 0, stream>>>(qb, kb, vT, bias, mask, opre);
    // output projection
    k_proj_gemm<<<dim3(MT2_N, DIM / 64), 32, 0, stream>>>(opre, wprojT, proj_b, out);
}
